// MyUNet_64759516889242
// MI455X (gfx1250) — compile-verified, matrix operand order checked by reading, not run on hardware
//
#include <hip/hip_runtime.h>
#include <hip/hip_bf16.h>
#include <math.h>

typedef __attribute__((ext_vector_type(16))) _Float16 v16h;
typedef __attribute__((ext_vector_type(8)))  float    v8f;

#define CONV_MAX_K   576   // max Cin*9  (u2 layer: 64*9)
#define CONV_MAX_CIN 64

// ---------------------------------------------------------------------------
// 3x3 SAME conv + bias + eval-BN + ReLU as implicit GEMM on WMMA.
// One wave (32 lanes) produces a 16(outC) x 16(pixels-in-a-row) tile.
//
// LDS panels sA/sB are staged in *fragment order*: [chunk][lane][16 halves],
// so the hot loop is just two 32B LDS vector loads + one v_wmma per K-chunk.
//
// Fragment K-mapping for V_WMMA_F32_16X16X32_F16 (ISA 7.12.2):
//   lane<16 (half 0): K = e<8 ? e   : e+8     (K in {0..7, 16..23})
//   lane>=16(half 1): K = e<8 ? e+8 : e+16    (K in {8..15, 24..31})
// Inverse (used at pack time): half=(k>>3)&1 ; e=(k&7)|((k&16)>>1).
// A: lane = half*16 + M (M=out channel). B: lane = half*16 + N (N=pixel).
// ---------------------------------------------------------------------------
__global__ __launch_bounds__(32) void conv3x3_bn_relu_wmma(
    const float* __restrict__ in1, int C1,
    const float* __restrict__ in2, int C2,
    const float* __restrict__ w,
    const float* __restrict__ bias,
    const float* __restrict__ gamma,
    const float* __restrict__ beta,
    float* __restrict__ out,
    int Cout, int H, int W)
{
    __shared__ alignas(32) _Float16 sA[CONV_MAX_K * 16];   // fragment-ordered A
    __shared__ alignas(32) _Float16 sB[CONV_MAX_K * 16];   // fragment-ordered B
    __shared__ _Float16 sI[CONV_MAX_CIN * 54];             // 3 x 18 halo per cin

    const int lane  = threadIdx.x;          // 0..31
    const int half  = lane >> 4;            // 0 or 1
    const int n16   = lane & 15;            // pixel column / out-channel index
    const int tilesPerRow = W >> 4;
    const int y     = blockIdx.x / tilesPerRow;
    const int x0    = (blockIdx.x % tilesPerRow) << 4;
    const int ocBase = blockIdx.y << 4;
    const int b     = blockIdx.z;
    const int Cin   = C1 + C2;
    const int K     = Cin * 9;
    const int Kpad  = (K + 31) & ~31;

    __builtin_prefetch(w, 0, 1);  // global_prefetch_b8 for the weight panel

    // ---- cooperative input halo load: rows y-1..y+1, cols x0-1..x0+16 ----
    const int nIn = Cin * 54;
    for (int i = lane; i < nIn; i += 32) {
        const int c  = i / 54;
        const int r  = i - c * 54;
        const int ry = r / 18;
        const int rx = r - ry * 18;
        const int yy = y + ry - 1;
        const int xx = x0 + rx - 1;
        float v = 0.0f;
        if (yy >= 0 && yy < H && xx >= 0 && xx < W) {
            if (c < C1) v = in1[((b * C1 + c) * H + yy) * W + xx];
            else        v = in2[((b * C2 + (c - C1)) * H + yy) * W + xx];
        }
        sI[i] = (_Float16)v;
    }

    // ---- pack A (weights) straight into fragment order, zero padded ----
    for (int i = lane; i < 16 * Kpad; i += 32) {
        const int k  = i >> 4;
        const int oc = i & 15;
        float v = 0.0f;
        const int gc = ocBase + oc;
        if (k < K && gc < Cout) v = w[gc * K + k];
        const int kp    = k & 31;
        const int halfk = (kp >> 3) & 1;
        const int e     = (kp & 7) | ((kp & 16) >> 1);
        sA[(((k >> 5) << 5) + halfk * 16 + oc) * 16 + e] = (_Float16)v;
    }
    __syncthreads();   // sB pack reads sI written by other lanes

    // ---- pack B (im2col) from halo into fragment order, zero padded ----
    for (int i = lane; i < 16 * Kpad; i += 32) {
        const int k = i >> 4;
        const int n = i & 15;
        _Float16 v = (_Float16)0.0f;
        if (k < K) {
            const int c  = k / 9;
            const int t  = k - c * 9;
            const int ky = t / 3;
            const int kx = t - ky * 3;
            v = sI[c * 54 + ky * 18 + n + kx];
        }
        const int kp    = k & 31;
        const int halfk = (kp >> 3) & 1;
        const int e     = (kp & 7) | ((kp & 16) >> 1);
        sB[(((k >> 5) << 5) + halfk * 16 + n) * 16 + e] = v;
    }
    __syncthreads();

    // ---- hot loop: 2x 32B LDS fragment loads + 1 WMMA per K-chunk ----
    v8f acc = {};
    const int nChunk = Kpad >> 5;
    const _Float16* pA = sA + lane * 16;
    const _Float16* pB = sB + lane * 16;
    for (int ch = 0; ch < nChunk; ++ch) {
        const v16h aF = *(const v16h*)(pA + ch * 512);
        const v16h bF = *(const v16h*)(pB + ch * 512);
        acc = __builtin_amdgcn_wmma_f32_16x16x32_f16(
                  false, aF, false, bF, (short)0, acc, false, false);
    }

    // ---- epilogue: bias -> BN(mean=0,var=1,eps=1e-5) -> ReLU ----
    const float s = 0.99999500003749981f;   // 1/sqrt(1 + 1e-5)
#pragma unroll
    for (int r = 0; r < 8; ++r) {
        const int oc = ocBase + r + half * 8;   // C layout: M = r + 8*half
        if (oc < Cout) {
            float v = acc[r] + bias[oc];
            v = gamma[oc] * (v * s) + beta[oc];
            v = v > 0.0f ? v : 0.0f;
            out[((b * Cout + oc) * H + y) * W + x0 + n16] = v;
        }
    }
}

// ---------------------------------------------------------------------------
// 2x2 max pool, stride 2.
// ---------------------------------------------------------------------------
__global__ void maxpool2x2(const float* __restrict__ in, float* __restrict__ out,
                           int BC, int Hi, int Wi)
{
    const int Ho = Hi >> 1, Wo = Wi >> 1;
    const long total = (long)BC * Ho * Wo;
    for (long idx = (long)blockIdx.x * blockDim.x + threadIdx.x; idx < total;
         idx += (long)gridDim.x * blockDim.x) {
        const int x = (int)(idx % Wo);
        long t = idx / Wo;
        const int yv = (int)(t % Ho);
        const long bc = t / Ho;
        const float* p = in + (bc * Hi + 2 * yv) * Wi + 2 * x;
        out[idx] = fmaxf(fmaxf(p[0], p[1]), fmaxf(p[Wi], p[Wi + 1]));
    }
}

// ---------------------------------------------------------------------------
// 2x bilinear upsample, align_corners = True.
// ---------------------------------------------------------------------------
__global__ void upsample2x_bilinear(const float* __restrict__ in, float* __restrict__ out,
                                    int BC, int Hi, int Wi)
{
    const int Ho = Hi * 2, Wo = Wi * 2;
    const float sy = (float)(Hi - 1) / (float)(Ho - 1);
    const float sx = (float)(Wi - 1) / (float)(Wo - 1);
    const long total = (long)BC * Ho * Wo;
    for (long idx = (long)blockIdx.x * blockDim.x + threadIdx.x; idx < total;
         idx += (long)gridDim.x * blockDim.x) {
        const int ox = (int)(idx % Wo);
        long t = idx / Wo;
        const int oy = (int)(t % Ho);
        const long bc = t / Ho;
        const float fy = oy * sy, fx = ox * sx;
        int y0 = (int)floorf(fy);
        int x0 = (int)floorf(fx);
        int y1 = min(y0 + 1, Hi - 1);
        int x1 = min(x0 + 1, Wi - 1);
        const float wy = fy - (float)y0, wx = fx - (float)x0;
        const float* p = in + bc * (long)Hi * Wi;
        const float v00 = p[(long)y0 * Wi + x0], v01 = p[(long)y0 * Wi + x1];
        const float v10 = p[(long)y1 * Wi + x0], v11 = p[(long)y1 * Wi + x1];
        const float r0 = v00 * (1.0f - wy) + v10 * wy;
        const float r1 = v01 * (1.0f - wy) + v11 * wy;
        out[idx] = r0 * (1.0f - wx) + r1 * wx;
    }
}

// ---------------------------------------------------------------------------
// FCAS: on one HxW channel (batch0, channel1), for each element v compute
// p=#(>v), n=#(==v), e=#(<v); interior pixels get (p*w0+b0+n*w1+b1+e*w2+b2)/3.
// Single block, channel staged in LDS (64*64*4 = 16KB), in-place update.
// ---------------------------------------------------------------------------
__global__ __launch_bounds__(256) void fcas_kernel(
    float* __restrict__ ch, const float* __restrict__ fw,
    const float* __restrict__ fb, int H, int W)
{
    __shared__ float s[4096];
    const int N = H * W;   // 4096
    for (int i = threadIdx.x; i < N; i += 256) s[i] = ch[i];
    __syncthreads();
    const float w0 = fw[0], w1 = fw[1], w2 = fw[2];
    const float b0 = fb[0], b1 = fb[1], b2 = fb[2];
    for (int i = threadIdx.x; i < N; i += 256) {
        const float v = s[i];
        int p = 0, n = 0, e = 0;
        for (int j = 0; j < N; ++j) {
            const float u = s[j];
            p += (u > v); n += (u == v); e += (u < v);
        }
        const int row = i / W, col = i - row * W;
        if (row >= 1 && row <= H - 2 && col >= 1 && col <= W - 2) {
            ch[i] = ((float)p * w0 + b0 + (float)n * w1 + b1 + (float)e * w2 + b2)
                    * (1.0f / 3.0f);
        }
    }
}

// ---------------------------------------------------------------------------
// Final 1x1 conv (4 -> 1 channel) + sigmoid.
// ---------------------------------------------------------------------------
__global__ void conv1x1_sigmoid(const float* __restrict__ in,
                                const float* __restrict__ w,
                                const float* __restrict__ bias,
                                float* __restrict__ out, int HW)
{
    const long total = 8L * HW;
    for (long idx = (long)blockIdx.x * blockDim.x + threadIdx.x; idx < total;
         idx += (long)gridDim.x * blockDim.x) {
        const int p  = (int)(idx % HW);
        const int bi = (int)(idx / HW);
        const float* q = in + (long)bi * 4 * HW + p;
        float acc = bias[0] + w[0] * q[0] + w[1] * q[(long)HW]
                            + w[2] * q[2L * HW] + w[3] * q[3L * HW];
        out[idx] = 1.0f / (1.0f + expf(-acc));
    }
}

// ---------------------------------------------------------------------------
extern "C" void kernel_launch(void* const* d_in, const int* in_sizes, int n_in,
                              void* d_out, int out_size, void* d_ws, size_t ws_size,
                              hipStream_t stream)
{
    (void)in_sizes; (void)n_in; (void)out_size; (void)ws_size;
    const float* x = (const float*)d_in[0];

    // Workspace arenas (floats).  Persistent skip tensors + two ping-pong temps.
    float* W  = (float*)d_ws;
    float* x1 = W;                          // 8* 8*512*512 = 16777216
    float* x2 = W + 16777216;               // 8*16*256*256 =  8388608
    float* x3 = W + 25165824;               // 8*32*128*128 =  4194304
    float* A  = W + 29360128;               // arena A: 16777216 floats
    float* Bz = W + 46137344;               // arena B:  8388608 floats
    // total: 54525952 floats = ~208 MB of d_ws

    auto conv = [&](const float* in1, int C1, const float* in2, int C2,
                    int pi, float* outp, int Cout, int H, int Wd) {
        dim3 g((unsigned)((Wd >> 4) * H), (unsigned)((Cout + 15) / 16), 8u);
        conv3x3_bn_relu_wmma<<<g, 32, 0, stream>>>(
            in1, C1, in2, C2,
            (const float*)d_in[pi], (const float*)d_in[pi + 1],
            (const float*)d_in[pi + 2], (const float*)d_in[pi + 3],
            outp, Cout, H, Wd);
    };
    auto grid1d = [](long total) { return (unsigned)((total + 255) / 256); };

    // Encoder
    conv(x, 3, nullptr, 0, 1, x1, 8, 512, 512);                           // x1
    maxpool2x2<<<grid1d(64L*256*256), 256, 0, stream>>>(x1, A, 8*8, 512, 512);
    conv(A, 8, nullptr, 0, 5, x2, 16, 256, 256);                          // x2
    maxpool2x2<<<grid1d(128L*128*128), 256, 0, stream>>>(x2, Bz, 8*16, 256, 256);
    conv(Bz, 16, nullptr, 0, 9, x3, 32, 128, 128);                        // x3
    maxpool2x2<<<grid1d(256L*64*64), 256, 0, stream>>>(x3, A, 8*32, 128, 128);
    conv(A, 32, nullptr, 0, 13, Bz, 32, 64, 64);                          // x4 in Bz

    // FCAS on batch 0, channel 1 of x4
    fcas_kernel<<<1, 256, 0, stream>>>(Bz + 4096,
        (const float*)d_in[31], (const float*)d_in[32], 64, 64);

    // Decoder (concats handled by channel-split conv inputs)
    upsample2x_bilinear<<<grid1d(256L*128*128), 256, 0, stream>>>(Bz, A, 8*32, 64, 64);
    conv(x3, 32, A, 32, 17, Bz, 16, 128, 128);                            // u2 in Bz
    upsample2x_bilinear<<<grid1d(128L*256*256), 256, 0, stream>>>(Bz, A, 8*16, 128, 128);
    conv(x2, 16, A, 16, 21, Bz, 8, 256, 256);                             // u3 in Bz
    upsample2x_bilinear<<<grid1d(64L*512*512), 256, 0, stream>>>(Bz, A, 8*8, 256, 256);
    conv(x1, 8, A, 8, 25, Bz, 4, 512, 512);                               // u4 in Bz

    conv1x1_sigmoid<<<grid1d(8L*512*512), 256, 0, stream>>>(
        Bz, (const float*)d_in[29], (const float*)d_in[30],
        (float*)d_out, 512 * 512);
}
